// SPLINE_LAYER_25933012533555
// MI455X (gfx1250) — matrix operand order repaired and checked by reading, not compiled, not run on hardware
//
#include <hip/hip_runtime.h>
#include <hip/hip_bf16.h>
#include <math.h>

typedef __attribute__((ext_vector_type(16))) _Float16 v16h;
typedef __attribute__((ext_vector_type(8)))  float    v8f;

#define M_ROWS 32768
#define NVC    64
#define NH     960            // hidden width
#define NPQ    960            // combined param width (640 pP + 320 pO)
#define TM     16             // rows per block (= one WMMA M tile)
#define NT1    60             // N-tiles (960/16)
#define KC1    2              // K-chunks GEMM1 (64/32, K padded 48->64)
#define KC2    30             // K-chunks GEMM2 (960/32)
#define NBINS  10
#define MBW_F  0.001f
#define PI_F   3.14159265358979323846f
#define SLOPE_F 1.3142616875182228f   // log1p(e) + 0.001

// ---------------- WMMA fragment helpers (wave32, 16x16x32 f16) ----------------

// B-matrix (32x16 f16): lane L -> N=L%16; half e<8 -> K=e+8*(L>>4), e>=8 -> K=16+(e-8)+8*(L>>4).
__device__ __forceinline__ int fragBK(int e, int lane) {
  return (e & 7) + ((lane >> 4) << 3) + ((e >> 3) << 4);
}
// Branch-free inverse of the A layout: K-in-chunk k (0..31) lives at
// lane (m + laneOfs), half e.
__device__ __forceinline__ void fragAslot(int k, int& laneOfs, int& e) {
  laneOfs = (k & 8) << 1;              // 16 iff k in [8,16) or [24,32)
  e = (k & 7) | ((k >> 1) & 8);        // +8 iff k >= 16
}

__device__ __forceinline__ v16h loadB_frag(const _Float16* base, int fragIdx, int lane) {
  // fragment-major packed: 32 lanes x 16 halves (32B per lane, aligned)
  return *(const v16h*)(base + ((size_t)fragIdx * 32 + lane) * 16);
}

__device__ __forceinline__ v8f wmma16(v16h a, v16h b, v8f c) {
  return __builtin_amdgcn_wmma_f32_16x16x32_f16(false, a, false, b, (short)0, c, false, false);
}

// GEMM1 A-fragments built straight from global x, branch-free.
// enc columns: [0..15]=cos(pi*xcP), [16..31]=sin(pi*xcP), [32..47]=xcO, [48..63]=0
// A layout: lane L (m=L&15, g8=(L>>4)*8), half e<8 -> K=e+g8 ; e>=8 -> K=16+(e-8)+g8.
__device__ __forceinline__ v16h encA0(const float* __restrict__ xr, int condBase, int lane) {
  int g8 = (lane >> 4) << 3;
  v16h r;
#pragma unroll
  for (int e = 0; e < 8; ++e) {        // K = e+g8 in [0,16): cos
    int j = e + g8;
    r[e] = (_Float16)__cosf(xr[condBase + 2 * j] * PI_F);
  }
#pragma unroll
  for (int e = 8; e < 16; ++e) {       // K = 16+(e-8)+g8 in [16,32): sin
    int j = (e - 8) + g8;
    r[e] = (_Float16)__sinf(xr[condBase + 2 * j] * PI_F);
  }
  return r;
}
__device__ __forceinline__ v16h encA32(const float* __restrict__ xr, int condBase, int lane) {
  int g8 = (lane >> 4) << 3;
  v16h r;
#pragma unroll
  for (int e = 0; e < 8; ++e) {        // K = 32+e+g8 in [32,48): xcO passthrough
    int j = e + g8;
    r[e] = (_Float16)xr[condBase + 2 * j + 1];
  }
#pragma unroll
  for (int e = 8; e < 16; ++e) r[e] = (_Float16)0.0f;   // K in [48,64): zero pad
  return r;
}

// ---------------- weight packing (f32 -> f16 fragment-major) ----------------

__global__ void pack_wh(const float* __restrict__ W, _Float16* __restrict__ dst) {
  int idx = blockIdx.x * blockDim.x + threadIdx.x;
  if (idx >= NT1 * KC1 * 512) return;
  int e = idx & 15;
  int lane = (idx >> 4) & 31;
  int kc = (idx >> 9) % KC1;
  int nt = idx / (KC1 * 512);
  int K = kc * 32 + fragBK(e, lane);
  int N = nt * 16 + (lane & 15);
  dst[idx] = (_Float16)((K < 48) ? W[K * NH + N] : 0.0f);   // W: 48 x 960
}

__global__ void pack_wpo(const float* __restrict__ WP, const float* __restrict__ WO,
                         _Float16* __restrict__ dst) {
  int idx = blockIdx.x * blockDim.x + threadIdx.x;
  if (idx >= NT1 * KC2 * 512) return;
  int e = idx & 15;
  int lane = (idx >> 4) & 31;
  int kc = (idx >> 9) % KC2;
  int nt = idx / (KC2 * 512);
  int K = kc * 32 + fragBK(e, lane);                         // K in [0,960)
  int N = nt * 16 + (lane & 15);                             // N in [0,960)
  float v = (N < 640) ? WP[K * 640 + N] : WO[K * 320 + (N - 640)];
  dst[idx] = (_Float16)v;
}

// ---------------- rational-quadratic spline (tfp math) ----------------

__device__ __forceinline__ float wrap2(float v) {  // mod(v,2)-1
  float m = fmodf(v, 2.0f);
  if (m < 0.0f) m += 2.0f;
  return m - 1.0f;
}

__device__ __forceinline__ float rqs_fwd1(float x, const _Float16* wr, const _Float16* hr,
                                          float& ladj) {
  float w[NBINS], h[NBINS];
  float mw = (float)wr[0], mh = (float)hr[0];
#pragma unroll
  for (int b = 1; b < NBINS; ++b) {
    mw = fmaxf(mw, (float)wr[b]);
    mh = fmaxf(mh, (float)hr[b]);
  }
  float sw = 0.f, sh = 0.f;
#pragma unroll
  for (int b = 0; b < NBINS; ++b) {
    w[b] = __expf((float)wr[b] - mw); sw += w[b];
    h[b] = __expf((float)hr[b] - mh); sh += h[b];
  }
  const float c = 2.0f - NBINS * MBW_F;
  float fw = c / sw, fh = c / sh;
#pragma unroll
  for (int b = 0; b < NBINS; ++b) { w[b] = w[b] * fw + MBW_F; h[b] = h[b] * fh + MBW_F; }

  int bin = 0;
  float xk = -1.f, yk = -1.f, wk = w[0], hk = h[0];
  float cx = -1.f, cy = -1.f;
#pragma unroll
  for (int b = 0; b < NBINS - 1; ++b) {
    cx += w[b]; cy += h[b];
    if (x >= cx) { bin = b + 1; xk = cx; yk = cy; wk = w[b + 1]; hk = h[b + 1]; }
  }
  float d0 = (bin == 0) ? 1.f : SLOPE_F;
  float d1 = (bin == NBINS - 1) ? 1.f : SLOPE_F;
  float th  = (x - xk) / wk;
  float sk  = hk / wk;
  float t1m = th * (1.f - th);
  float den = sk + (d0 + d1 - 2.f * sk) * t1m;
  float y   = yk + hk * (sk * th * th + d0 * t1m) / den;
  float dydx = sk * sk * (d1 * th * th + 2.f * sk * t1m + d0 * (1.f - th) * (1.f - th)) / (den * den);
  bool inr = (x >= -1.f) && (x <= 1.f);
  if (inr) ladj += __logf(dydx);
  return inr ? y : x;
}

// ---------------- fused coupling kernel ----------------
// transBase=0 (coupling A: conditioner = cols 32..63) or 32 (coupling B).
// xout may alias xin (each block only touches its own 16 rows; reads precede writes).

__global__ void __launch_bounds__(256) coupling_kernel(
    const float* __restrict__ xin, float* __restrict__ xout,
    const _Float16* __restrict__ whFrag, const float* __restrict__ bh,
    const _Float16* __restrict__ wpoFrag, const float* __restrict__ bP,
    const float* __restrict__ bO,
    const float* __restrict__ ladjIn, float* __restrict__ ladjOut, int transBase) {
  // hid activations stored A-FRAGMENT-MAJOR: [kc][lane][half] so GEMM2's
  // A-load is one aligned v16h (2x ds_load_b128), zero index math in the loop.
  __shared__ _Float16 __align__(32) hidA[KC2 * 32 * 16];   // 30 KB
  __shared__ _Float16 prm[TM * NPQ];                       // 30 KB
  __shared__ float ladjS[TM];

  const int t = threadIdx.x;
  const int lane = t & 31, wid = t >> 5;
  const int rowBase = blockIdx.x * TM;
  const int condBase = transBase ^ 32;

  if (t < TM) ladjS[t] = 0.f;

  // ---- GEMM1: enc(16x64) @ Wh(64x960) -> relu -> hidA (f16, frag-major) ----
  {
    const float* xr = xin + (size_t)(rowBase + (lane & 15)) * NVC;
    v16h a0 = encA0(xr, condBase, lane);
    v16h a1 = encA32(xr, condBase, lane);
#pragma unroll
    for (int i = 0; i < 8; ++i) {
      int nt = wid + i * 8;                       // wave-uniform
      if (nt < NT1) {
        v8f acc = {};
        v16h b0 = loadB_frag(whFrag, nt * KC1 + 0, lane);
        v16h b1 = loadB_frag(whFrag, nt * KC1 + 1, lane);
        acc = wmma16(a0, b0, acc);
        acc = wmma16(a1, b1, acc);
        int n = nt * 16 + (lane & 15);            // this lane's output column
        float bias = bh[n];
        int kc = n >> 5, k = n & 31;
        int laneOfs, e;
        fragAslot(k, laneOfs, e);                 // branch-free
        int rb = (lane >> 4) * 8;
        _Float16* dstBase = hidA + (size_t)(kc * 32 + laneOfs) * 16 + e;
#pragma unroll
        for (int r = 0; r < 8; ++r) {
          float v = fmaxf(acc[r] + bias, 0.f);
          dstBase[(size_t)(r + rb) * 16] = (_Float16)v;   // row m = r+rb
        }
      }
    }
  }
  __syncthreads();

  // ---- GEMM2: hid(16x960) @ [WP|WO](960x960) -> prm (f16) ----
  {
    const v16h* aFrag = (const v16h*)hidA;
    v8f acc[8] = {};
    for (int kc = 0; kc < KC2; ++kc) {
      v16h a = aFrag[kc * 32 + lane];             // 2x ds_load_b128
#pragma unroll
      for (int i = 0; i < 8; ++i) {
        int nt = wid + i * 8;                     // wave-uniform
        if (nt < NT1) {
          v16h b = loadB_frag(wpoFrag, nt * KC2 + kc, lane);
          acc[i] = wmma16(a, b, acc[i]);
        }
      }
    }
#pragma unroll
    for (int i = 0; i < 8; ++i) {
      int nt = wid + i * 8;
      if (nt < NT1) {
        int n = nt * 16 + (lane & 15);
        float bias = (n < 640) ? bP[n] : bO[n - 640];
        int rb = (lane >> 4) * 8;
#pragma unroll
        for (int r = 0; r < 8; ++r)
          prm[(r + rb) * NPQ + n] = (_Float16)(acc[i][r] + bias);
      }
    }
  }
  __syncthreads();

  // ---- spline epilogue: thread (row, elem) ----
  {
    int row = t >> 4, j = t & 15;
    const _Float16* pr = prm + row * NPQ;
    const float* xr = xin + (size_t)(rowBase + row) * NVC;
    float* yr = xout + (size_t)(rowBase + row) * NVC;
    float ladj = 0.f;

    float y = xr[transBase + 2 * j];
    // spline 0 (shift 0)
    y = wrap2(y + 1.0f);
    y = rqs_fwd1(y, pr + 0   + j * NBINS, pr + 160 + j * NBINS, ladj);
    y = wrap2(y + 1.0f);
    // spline 1 (shift 1)
    y = wrap2(y + 2.0f);
    y = rqs_fwd1(y, pr + 320 + j * NBINS, pr + 480 + j * NBINS, ladj);
    y = wrap2(y + 0.0f);
    // xO spline
    float yo = xr[transBase + 2 * j + 1];
    yo = rqs_fwd1(yo, pr + 640 + j * NBINS, pr + 800 + j * NBINS, ladj);

    atomicAdd(&ladjS[row], ladj);      // ds_add_f32

    yr[transBase + 2 * j]     = y;
    yr[transBase + 2 * j + 1] = yo;
    yr[condBase + 2 * j]      = xr[condBase + 2 * j];
    yr[condBase + 2 * j + 1]  = xr[condBase + 2 * j + 1];
  }
  __syncthreads();

  if (t < TM) {
    int g = rowBase + t;
    float l = ladjS[t];
    ladjOut[g] = ladjIn ? (ladjIn[g] + l) : l;
  }
}

// ---------------- host launcher ----------------

extern "C" void kernel_launch(void* const* d_in, const int* in_sizes, int n_in,
                              void* d_out, int out_size, void* d_ws, size_t ws_size,
                              hipStream_t stream) {
  (void)in_sizes; (void)n_in; (void)out_size; (void)ws_size;

  const float* x   = (const float*)d_in[0];
  const float* WAh = (const float*)d_in[1];
  const float* bAh = (const float*)d_in[2];
  const float* WAP = (const float*)d_in[3];
  const float* bAP = (const float*)d_in[4];
  const float* WAO = (const float*)d_in[5];
  const float* bAO = (const float*)d_in[6];
  const float* WBh = (const float*)d_in[7];
  const float* bBh = (const float*)d_in[8];
  const float* WBP = (const float*)d_in[9];
  const float* bBP = (const float*)d_in[10];
  const float* WBO = (const float*)d_in[11];
  const float* bBO = (const float*)d_in[12];

  char* ws = (char*)d_ws;
  size_t off = 0;
  auto carve = [&](size_t bytes) -> char* {
    char* p = ws + off;
    off = (off + bytes + 255) & ~(size_t)255;
    return p;
  };
  float*    ladj1 = (float*)carve((size_t)M_ROWS * sizeof(float));
  _Float16* whA   = (_Float16*)carve((size_t)NT1 * KC1 * 512 * 2);
  _Float16* wpoA  = (_Float16*)carve((size_t)NT1 * KC2 * 512 * 2);
  _Float16* whB   = (_Float16*)carve((size_t)NT1 * KC1 * 512 * 2);
  _Float16* wpoB  = (_Float16*)carve((size_t)NT1 * KC2 * 512 * 2);

  pack_wh<<<(NT1 * KC1 * 512 + 255) / 256, 256, 0, stream>>>(WAh, whA);
  pack_wh<<<(NT1 * KC1 * 512 + 255) / 256, 256, 0, stream>>>(WBh, whB);
  pack_wpo<<<(NT1 * KC2 * 512 + 255) / 256, 256, 0, stream>>>(WAP, WAO, wpoA);
  pack_wpo<<<(NT1 * KC2 * 512 + 255) / 256, 256, 0, stream>>>(WBP, WBO, wpoB);

  float* xoutF   = (float*)d_out;                 // (M, 64)
  float* ladjOut = xoutF + (size_t)M_ROWS * NVC;  // (M, 1)

  // Coupling A: x -> d_out x-region, ladj1 in ws
  coupling_kernel<<<M_ROWS / TM, 256, 0, stream>>>(
      x, xoutF, whA, bAh, wpoA, bAP, bAO, nullptr, ladj1, 0);
  // Coupling B: in-place on d_out x-region, ladjOut = ladj1 + ladj2
  coupling_kernel<<<M_ROWS / TM, 256, 0, stream>>>(
      xoutF, xoutF, whB, bBh, wpoB, bBP, bBO, ladj1, ladjOut, 32);
}
